// DeformableConv2d_32435593020128
// MI455X (gfx1250) — compile-verified, matrix operand order checked by reading, not run on hardware
//
#include <hip/hip_runtime.h>
#include <math.h>

// ---------------------------------------------------------------------------
// Modulated Deformable Conv v2 on gfx1250 (MI455X), bf16 WMMA path.
// B=4, C=256, H=W=Ho=Wo=64, O=256, K=3 (K2=9), PAD=1, STRIDE=1.
// Main GEMM: 19.3 GFLOP, everything L2-resident -> compute-bound on WMMA.
// A-tiles are DMA'd to LDS with the Tensor Data Mover (double-buffered),
// B-tiles are bilinear-sampled on the fly (corner table precomputed once
// per (b,k,p) and reused across all 256 channels).
// ---------------------------------------------------------------------------

typedef __bf16 bf16_t;
typedef bf16_t bf16x4  __attribute__((ext_vector_type(4)));
typedef bf16_t bf16x8  __attribute__((ext_vector_type(8)));
typedef bf16_t bf16x16 __attribute__((ext_vector_type(16)));
typedef float  f32x8   __attribute__((ext_vector_type(8)));
typedef unsigned int v4u __attribute__((ext_vector_type(4)));
typedef int v4i __attribute__((ext_vector_type(4)));
typedef int v8i __attribute__((ext_vector_type(8)));

#define NB    4
#define NC    256
#define HW    4096      // 64*64
#define NO    256
#define K2    9
#define CCHNK 8         // 2304 = 9 taps * 8 chunks * 32 ch

// Workspace byte offsets (all 256-aligned). Total ~7.5 MB.
#define WS_WMAIN   0u          // bf16 [9][8][256][32]   = 1,179,648 B
#define WS_WOM     1179648u    // bf16 [9][8][32][32]    =   147,456 B
#define WS_OFF     1327104u    // f32  [B][18][4096]     = 1,179,648 B
#define WS_MASK    2506752u    // f32  [B][9][4096]      =   589,824 B
#define WS_MIDX    3096576u    // int4 [B][9][4096]      = 2,359,296 B
#define WS_MWGT    5455872u    // float4 [B][9][4096]    = 2,359,296 B

#if __has_builtin(__builtin_amdgcn_tensor_load_to_lds) && \
    __has_builtin(__builtin_amdgcn_s_wait_tensorcnt)
#define USE_TDM 1
#else
#define USE_TDM 0
#endif

#if USE_TDM
// Issue one TDM descriptor: contiguous 16 KB (8192 x 2B) global -> LDS.
// D# group0: count=1, lds_addr, 57b global_addr, type=2 ("image").
// D# group1: data_size=1(2B), tensor_dim0=tile_dim0=8192, tensor_dim1=1,
//            tile_dim1=1, dim0_stride=8192 (bit-packed per ISA 8.4).
__device__ __forceinline__ void tdm_load_a16k(unsigned lds_off, const void* gsrc)
{
    unsigned long long ga =
        ((unsigned long long)(size_t)gsrc) & 0x01FFFFFFFFFFFFFFull;
    v4u g0 = { 1u, lds_off, (unsigned)(ga & 0xFFFFFFFFu),
               (unsigned)(ga >> 32) | 0x80000000u };
    v8i g1 = { 0x00010000, 0x20000000, 0x00010000, 0x20000000,
               1, 8192, 0x20000000, 0 };
    v4i z4 = { 0, 0, 0, 0 };
#if defined(__clang_major__) && __clang_major__ >= 23
    v8i z8 = { 0, 0, 0, 0, 0, 0, 0, 0 };
    __builtin_amdgcn_tensor_load_to_lds(g0, g1, z4, z4, z8, 0);
#else
    __builtin_amdgcn_tensor_load_to_lds(g0, g1, z4, z4, 0);
#endif
}
#endif

// ---------------------------------------------------------------------------
// Kernel 1: pack weights into WMMA-friendly bf16 tile layouts.
//   Wmain[((k*8+cc)*256+o)*32 + c32] = weight[o][cc*32+c32][k]
//   Wom  [((k*8+cc)*32 +r)*32 + c32] = {offset_w | mod_w | 0}[r][c][k]
// ---------------------------------------------------------------------------
__global__ __launch_bounds__(256) void pack_weights(
    const float* __restrict__ offset_w,   // [18,256,3,3]
    const float* __restrict__ mod_w,      // [9,256,3,3]
    const float* __restrict__ w_main,     // [256,256,3,3]
    bf16_t* __restrict__ Wom,
    bf16_t* __restrict__ Wmain)
{
    int d = blockIdx.x * 256 + threadIdx.x;
    const int MAIN_N = 9 * 8 * 256 * 32;          // 589824 == 9<<16
    const int OM_N   = 9 * 8 * 32 * 32;           // 73728  == 9<<13
    if (d < MAIN_N) {
        int c32 = d & 31;
        int o   = (d >> 5) & 255;
        int cc  = (d >> 13) & 7;
        int k   = d >> 16;
        int c   = cc * 32 + c32;
        Wmain[d] = (bf16_t)w_main[(o * NC + c) * K2 + k];
    } else if (d < MAIN_N + OM_N) {
        int e   = d - MAIN_N;
        int c32 = e & 31;
        int r   = (e >> 5) & 31;
        int cc  = (e >> 10) & 7;
        int k   = e >> 13;
        int c   = cc * 32 + c32;
        float v = 0.0f;
        if (r < 18)      v = offset_w[(r * NC + c) * K2 + k];
        else if (r < 27) v = mod_w[((r - 18) * NC + c) * K2 + k];
        Wom[e] = (bf16_t)v;
    }
}

// ---------------------------------------------------------------------------
// Kernel 2: offset/mask conv as WMMA implicit GEMM.
//   M=32 (27 real rows), N=128 pixels/block, K=2304.
//   256 threads = 8 waves: wave -> (m-sub = w&1, n-block = w>>1, 2 subtiles).
// ---------------------------------------------------------------------------
__global__ __launch_bounds__(256) void om_conv(
    const float* __restrict__ x,          // [B,256,64,64]
    const bf16_t* __restrict__ Wom,       // [9][8][32][32]
    const float* __restrict__ offb,       // [18]
    const float* __restrict__ modb,       // [9]
    float* __restrict__ off,              // [B,18,4096]
    float* __restrict__ maskArr)          // [B,9,4096]
{
    __shared__ alignas(32) bf16_t As[32 * 32];    // 2 KB
    __shared__ alignas(32) bf16_t Bt[128 * 32];   // 8 KB, [n][k32]

    const int b  = blockIdx.x >> 5;
    const int p0 = (blockIdx.x & 31) << 7;        // 128 pixels
    const int t    = threadIdx.x;
    const int lane = t & 31;
    const int wave = t >> 5;

    f32x8 acc[2] = {};
    const float* xb = x + (size_t)b * NC * HW;

    const int sp   = t & 127;          // pixel within tile (sampling role)
    const int chal = (t >> 7) * 16;    // channel half (sampling role)

    for (int k = 0; k < K2; ++k) {
        const int ky = k / 3 - 1;
        const int kx = k % 3 - 1;
        const int py = ((p0 + sp) >> 6) + ky;
        const int px = (sp & 63) + kx;
        const bool valid = (py >= 0) & (py < 64) & (px >= 0) & (px < 64);
        const int idx = valid ? (py * 64 + px) : 0;     // idx always safe
        const float scale = valid ? 1.0f : 0.0f;        // branch-free zero pad

        for (int cc = 0; cc < CCHNK; ++cc) {
            const int chunk = k * 8 + cc;
            __syncthreads();
            // stage A tile: 2 KB contiguous
            ((uint2*)As)[t] = ((const uint2*)(Wom + (size_t)chunk * 1024))[t];
            __builtin_prefetch(Wom + (size_t)(chunk + 1 < 72 ? chunk + 1 : chunk) * 1024
                                   + (size_t)(t & 15) * 64, 0, 1);
            // sample B tile (im2col, zero pad): 16 channels / thread,
            // packed bf16x8 stores -> ds_store_b128
            {
                const float* xc = xb + (size_t)(cc * 32 + chal) * HW + idx;
                bf16x8 pk0, pk1;
                #pragma unroll
                for (int i = 0; i < 8; ++i)
                    pk0[i] = (bf16_t)(xc[(size_t)i * HW] * scale);
                #pragma unroll
                for (int i = 0; i < 8; ++i)
                    pk1[i] = (bf16_t)(xc[(size_t)(i + 8) * HW] * scale);
                *(bf16x8*)(Bt + sp * 32 + chal)     = pk0;
                *(bf16x8*)(Bt + sp * 32 + chal + 8) = pk1;
            }
            __syncthreads();
            // compute
            const int o0 = (wave & 1) * 16;
            bf16x16 afrag = *(const bf16x16*)(As + (o0 + (lane & 15)) * 32 + (lane >> 4) * 16);
            const int nbase = (wave >> 1) * 32;
            #pragma unroll
            for (int s = 0; s < 2; ++s) {
                bf16x16 bfrag = *(const bf16x16*)(Bt + (nbase + s * 16 + (lane & 15)) * 32 + (lane >> 4) * 16);
                acc[s] = __builtin_amdgcn_wmma_f32_16x16x32_bf16(
                    false, afrag, false, bfrag, (short)0, acc[s], false, false);
            }
        }
    }

    // epilogue: bias + sigmoid routing
    const int obase = (wave & 1) * 16 + (lane >> 4) * 8;
    const int cbase = p0 + (wave >> 1) * 32 + (lane & 15);
    #pragma unroll
    for (int s = 0; s < 2; ++s) {
        const int p = cbase + s * 16;
        #pragma unroll
        for (int v = 0; v < 8; ++v) {
            const int o = obase + v;
            float a = acc[s][v];
            if (o < 18) {
                off[((size_t)b * 18 + o) * HW + p] = a + offb[o];
            } else if (o < 27) {
                float z = a + modb[o - 18];
                maskArr[((size_t)b * 9 + (o - 18)) * HW + p] = 2.0f / (1.0f + __expf(-z));
            }
        }
    }
}

// ---------------------------------------------------------------------------
// Kernel 3: per-(b,k,p) bilinear corner table, mask folded into weights.
// ---------------------------------------------------------------------------
__global__ __launch_bounds__(256) void sample_prep(
    const float* __restrict__ off,        // [B,18,4096]
    const float* __restrict__ maskArr,    // [B,9,4096]
    int4*  __restrict__ midx,             // [B*9*4096]
    float4* __restrict__ mwgt)
{
    int gid = blockIdx.x * 256 + threadIdx.x;
    if (gid >= NB * K2 * HW) return;
    int p = gid & (HW - 1);
    int k = (gid >> 12) % K2;
    int b = gid / (K2 * HW);

    float dy = off[((size_t)b * 18 + 2 * k) * HW + p];
    float dx = off[((size_t)b * 18 + 2 * k + 1) * HW + p];
    float m  = maskArr[(size_t)gid];

    float py = dy + (float)(k / 3) + (float)(p >> 6) - 1.0f;
    float px = dx + (float)(k % 3) + (float)(p & 63) - 1.0f;
    float y0f = floorf(py), x0f = floorf(px);
    int y0 = (int)y0f, x0 = (int)x0f;
    float fy = py - y0f, fx = px - x0f;

    int   yy[2]  = { y0, y0 + 1 }, xx[2] = { x0, x0 + 1 };
    float wys[2] = { 1.0f - fy, fy }, wxs[2] = { 1.0f - fx, fx };

    int4 id; float4 wv;
    int*   ip = &id.x;
    float* wp = &wv.x;
    #pragma unroll
    for (int j = 0; j < 2; ++j) {
        #pragma unroll
        for (int i = 0; i < 2; ++i) {
            int y = yy[j], xq = xx[i];
            bool ok = (y >= 0) & (y < 64) & (xq >= 0) & (xq < 64);
            int yc = y < 0 ? 0 : (y > 63 ? 63 : y);
            int xc = xq < 0 ? 0 : (xq > 63 ? 63 : xq);
            ip[j * 2 + i] = yc * 64 + xc;
            wp[j * 2 + i] = ok ? (wys[j] * wxs[i] * m) : 0.0f;
        }
    }
    midx[gid] = id;
    mwgt[gid] = wv;
}

// ---------------------------------------------------------------------------
// Kernel 4: fused deformable-sample + main GEMM.
//   Block: 512 threads = 16 waves. Tile M=256 (all out channels), N=64
//   (one image row). K loop: 9 taps x 8 chunks of 32 channels.
//   A-tiles (16 KB) double-buffered via TDM tensor_load_to_lds; DMA of
//   chunk n+1 overlaps sampling + WMMA of chunk n (s_wait_tensorcnt 1).
// ---------------------------------------------------------------------------
__global__ __launch_bounds__(512) void dcn_main(
    const float* __restrict__ x,          // [B,256,4096]
    const bf16_t* __restrict__ Wb,        // [9][8][256][32]
    const int4*  __restrict__ midx,       // [B][9][4096]
    const float4* __restrict__ mwgt,
    float* __restrict__ out)              // [B,256,4096]
{
    __shared__ alignas(32) bf16_t As[2][8192];    // 2 x 16 KB (double buffer)
    __shared__ alignas(32) bf16_t Bt[64 * 32];    // 4 KB, [n][k32]
    __shared__ alignas(16) int    sIdx[64][4];
    __shared__ alignas(16) float  sW[64][4];

    const int b  = blockIdx.x >> 6;
    const int p0 = (blockIdx.x & 63) << 6;        // one image row (64 px)
    const int t    = threadIdx.x;
    const int lane = t & 31;
    const int wave = t >> 5;                      // 0..15

    f32x8 acc[4] = {};
    const float* xb = x + (size_t)b * NC * HW;

    const int sp = t & 63;        // pixel (sampling role)
    const int sc = (t >> 6) * 4;  // first of 4 channels (sampling role)

#if USE_TDM
    if (wave == 0)
        tdm_load_a16k((unsigned)(size_t)&As[0][0], Wb);
#endif

    for (int k = 0; k < K2; ++k) {
        __syncthreads();   // previous tap fully consumed (meta reusable)
        if (t < 64) {
            ((int4*)sIdx[t])[0] = midx[((size_t)b * K2 + k) * HW + p0 + t];
        } else if (t < 128) {
            int j = t - 64;
            ((float4*)sW[j])[0] = mwgt[((size_t)b * K2 + k) * HW + p0 + j];
        }
        __syncthreads();

        for (int cc = 0; cc < CCHNK; ++cc) {
            const int chunk = k * 8 + cc;
            __syncthreads();       // chunk-1 compute done: Bt writable, spare
                                   // A buffer free for the next DMA
#if USE_TDM
            if (wave == 0 && chunk + 1 < 72)
                tdm_load_a16k((unsigned)(size_t)&As[(chunk + 1) & 1][0],
                              Wb + (size_t)(chunk + 1) * 8192);
#else
            {
                const float4* src = (const float4*)(Wb + (size_t)chunk * 8192);
                float4* dst = (float4*)&As[chunk & 1][0];
                dst[t]       = src[t];
                dst[t + 512] = src[t + 512];
            }
#endif
            // sample B tile: 4 channels per thread, 4-corner blend,
            // packed bf16x4 store -> ds_store_b64
            {
                const int i0 = sIdx[sp][0], i1 = sIdx[sp][1];
                const int i2 = sIdx[sp][2], i3 = sIdx[sp][3];
                const float w0 = sW[sp][0], w1 = sW[sp][1];
                const float w2 = sW[sp][2], w3 = sW[sp][3];
                const float* xc = xb + (size_t)(cc * 32 + sc) * HW;
                bf16x4 pk;
                #pragma unroll
                for (int i = 0; i < 4; ++i) {
                    const float* xp = xc + (size_t)i * HW;
                    pk[i] = (bf16_t)(w0 * xp[i0] + w1 * xp[i1] +
                                     w2 * xp[i2] + w3 * xp[i3]);
                }
                *(bf16x4*)(Bt + sp * 32 + sc) = pk;
            }
#if USE_TDM
            if (wave == 0) {
                if (chunk < 71) __builtin_amdgcn_s_wait_tensorcnt(1);
                else            __builtin_amdgcn_s_wait_tensorcnt(0);
            }
#endif
            __syncthreads();
            // compute: each wave owns 16 output rows x 64 pixels
            const bf16_t* Ac = &As[chunk & 1][0];
            bf16x16 afrag = *(const bf16x16*)(Ac + (wave * 16 + (lane & 15)) * 32 + (lane >> 4) * 16);
            #pragma unroll
            for (int s = 0; s < 4; ++s) {
                bf16x16 bfrag = *(const bf16x16*)(Bt + (s * 16 + (lane & 15)) * 32 + (lane >> 4) * 16);
                acc[s] = __builtin_amdgcn_wmma_f32_16x16x32_bf16(
                    false, afrag, false, bfrag, (short)0, acc[s], false, false);
            }
        }
    }

    // epilogue: D layout -> o = 16*wave + v + 8*(lane>>4); p = p0 + 16*s + (lane&15)
    const int obase = wave * 16 + (lane >> 4) * 8;
    float* ob = out + (size_t)b * NO * HW;
    #pragma unroll
    for (int s = 0; s < 4; ++s) {
        const int p = p0 + s * 16 + (lane & 15);
        #pragma unroll
        for (int v = 0; v < 8; ++v) {
            ob[(size_t)(obase + v) * HW + p] = acc[s][v];
        }
    }
}

// ---------------------------------------------------------------------------
extern "C" void kernel_launch(void* const* d_in, const int* in_sizes, int n_in,
                              void* d_out, int out_size, void* d_ws, size_t ws_size,
                              hipStream_t stream) {
    const float* x        = (const float*)d_in[0];
    const float* offset_w = (const float*)d_in[1];
    const float* offset_b = (const float*)d_in[2];
    const float* mod_w    = (const float*)d_in[3];
    const float* mod_b    = (const float*)d_in[4];
    const float* weight   = (const float*)d_in[5];
    float* out = (float*)d_out;

    char* ws = (char*)d_ws;
    bf16_t* Wmain   = (bf16_t*)(ws + WS_WMAIN);
    bf16_t* Wom     = (bf16_t*)(ws + WS_WOM);
    float*  off     = (float*)(ws + WS_OFF);
    float*  maskArr = (float*)(ws + WS_MASK);
    int4*   midx    = (int4*)(ws + WS_MIDX);
    float4* mwgt    = (float4*)(ws + WS_MWGT);

    // 1) pack weights to bf16 tiled layouts (663552 elements)
    pack_weights<<<2593, 256, 0, stream>>>(offset_w, mod_w, weight, Wom, Wmain);
    // 2) offset + modulation conv (WMMA implicit GEMM, M=32)
    om_conv<<<NB * (HW / 128), 256, 0, stream>>>(x, Wom, offset_b, mod_b, off, maskArr);
    // 3) bilinear corner/weight table, mask folded in
    sample_prep<<<(NB * K2 * HW) / 256, 256, 0, stream>>>(off, maskArr, midx, mwgt);
    // 4) fused deformable sample + main GEMM (WMMA + TDM double buffering)
    dcn_main<<<NB * (HW / 64), 512, 0, stream>>>(x, Wmain, midx, mwgt, out);
}
